// MetaPathGNN_2430951489548
// MI455X (gfx1250) — matrix operand order repaired and checked by reading, not compiled, not run on hardware
//
#include <hip/hip_runtime.h>

typedef __attribute__((ext_vector_type(16))) __bf16 v16bf;
typedef __attribute__((ext_vector_type(8)))  __bf16 v8bf;
typedef __attribute__((ext_vector_type(8)))  float  v8f;

#define HDIM 128
#define LDS_STRIDE 136   // 272B row pitch: 16B-aligned, bank-skewed

__device__ __forceinline__ unsigned short f2bf(float f) {
    __bf16 b = (__bf16)f;
    return __builtin_bit_cast(unsigned short, b);
}

__device__ __forceinline__ void atomic_add_f32(float* p, float v) {
    // guaranteed HW FP32 atomic (no CAS loop)
    asm volatile("global_atomic_add_f32 %0, %1, off" :: "v"(p), "v"(v) : "memory");
}

// ---------------------------------------------------------------- zero agg
__global__ void zero_f32x4(float* __restrict__ p, long long n4) {
    long long i = (long long)blockIdx.x * blockDim.x + threadIdx.x;
    if (i < n4) reinterpret_cast<float4*>(p)[i] = make_float4(0.f, 0.f, 0.f, 0.f);
}

// ------------------------------------------------- edge scatter: agg[dst]+=h[src]
// one wave per edge; each lane handles a float4 (128 floats / 32 lanes)
__global__ void scatter_add(const int* __restrict__ edges, int E,
                            const float* __restrict__ h, float* __restrict__ agg) {
    long long idx = (long long)blockIdx.x * blockDim.x + threadIdx.x;
    long long e = idx >> 5;
    if (e >= E) return;
    int q = (int)(idx & 31) * 4;
    int dst = edges[e];
    int src = edges[(long long)E + e];
    const float4 v = *reinterpret_cast<const float4*>(h + (size_t)src * HDIM + q);
    float* a = agg + (size_t)dst * HDIM + q;
    atomic_add_f32(a + 0, v.x);
    atomic_add_f32(a + 1, v.y);
    atomic_add_f32(a + 2, v.z);
    atomic_add_f32(a + 3, v.w);
}

// -------------------------------------- weight prep: bf16 transpose + bias sums
// grid: 64 x 256 = 16384 threads, one per (k,c)
__global__ void prep_weights(const float* __restrict__ wl1, const float* __restrict__ w01,
                             const float* __restrict__ w11,
                             const float* __restrict__ bl1, const float* __restrict__ b01,
                             const float* __restrict__ b11,
                             const float* __restrict__ wl0, const float* __restrict__ w00,
                             const float* __restrict__ w10,
                             const float* __restrict__ bl0, const float* __restrict__ b00,
                             const float* __restrict__ b10,
                             const float* __restrict__ wout,
                             unsigned short* __restrict__ wlT1, unsigned short* __restrict__ wsT1,
                             unsigned short* __restrict__ wlT0, unsigned short* __restrict__ wsT0,
                             unsigned short* __restrict__ woutT,
                             float* __restrict__ bias1, float* __restrict__ bias2) {
    int idx = blockIdx.x * blockDim.x + threadIdx.x;   // 0..16383
    int k = idx >> 7, c = idx & 127;
    int s = k * HDIM + c;          // row-major source
    int d = c * HDIM + k;          // column-major (transposed) dest
    wlT1[d]  = f2bf(wl1[s]);
    wsT1[d]  = f2bf(w01[s] + w11[s]);
    wlT0[d]  = f2bf(wl0[s]);
    wsT0[d]  = f2bf(w00[s] + w10[s]);
    woutT[d] = f2bf(wout[s]);
    if (idx < HDIM) {
        bias1[idx] = bl1[idx] + b01[idx] + b11[idx];
        bias2[idx] = bl0[idx] + b00[idx] + b10[idx];
    }
}

// ---------------------------------------------------------------- fused GEMM tile
// out[r0:r0+16, :] = act( A1@W1 (+ A2@W2) + bias ), K = N = 128
// 256 threads = 8 wave32; wave w owns 16 output columns [16w, 16w+16)
// HAS2/RELU are compile-time so the K-loop is branch-free; all fragment loads
// are hoisted ahead of the WMMA chain (one wait, 8 back-to-back WMMAs).
template <bool HAS2, bool RELU>
__global__ __launch_bounds__(256)
void gemm_tile(const float* __restrict__ A1, const unsigned short* __restrict__ W1u,
               const float* __restrict__ A2, const unsigned short* __restrict__ W2u,
               const float* __restrict__ bias, float* __restrict__ out) {
    __shared__ __bf16 lds1[16 * LDS_STRIDE];
    __shared__ __bf16 lds2[HAS2 ? 16 * LDS_STRIDE : 1];

    const int tid  = threadIdx.x;
    const int row0 = blockIdx.x * 16;

    // stage 16x128 fp32 rows -> bf16 LDS (512 float4 chunks / matrix, 2 per thread)
#pragma unroll
    for (int i = 0; i < 2; ++i) {
        int cidx = tid + i * 256;              // 0..511
        int r  = cidx >> 5;
        int c4 = (cidx & 31) * 4;
        float4 v = *reinterpret_cast<const float4*>(A1 + (size_t)(row0 + r) * HDIM + c4);
        __bf16* d1 = &lds1[r * LDS_STRIDE + c4];
        d1[0] = (__bf16)v.x; d1[1] = (__bf16)v.y; d1[2] = (__bf16)v.z; d1[3] = (__bf16)v.w;
        if constexpr (HAS2) {
            float4 u = *reinterpret_cast<const float4*>(A2 + (size_t)(row0 + r) * HDIM + c4);
            __bf16* d2 = &lds2[r * LDS_STRIDE + c4];
            d2[0] = (__bf16)u.x; d2[1] = (__bf16)u.y; d2[2] = (__bf16)u.z; d2[3] = (__bf16)u.w;
        }
    }
    __syncthreads();

    const int lane = tid & 31;
    const int wv   = tid >> 5;          // 0..7 -> column block
    const int m    = lane & 15;         // A row / C-N index held by this lane
    const int kh   = lane >> 4;         // K-half selector
    const int col  = wv * 16 + m;

    const __bf16* Wt1 = reinterpret_cast<const __bf16*>(W1u);
    const __bf16* Wt2 = reinterpret_cast<const __bf16*>(W2u);
    const __bf16* a1row = &lds1[m * LDS_STRIDE];
    const __bf16* a2row = &lds2[m * LDS_STRIDE];

    // ---- hoisted fragment loads -------------------------------------------
    v16bf aF1[4], bF1[4];
    v16bf aF2[4], bF2[4];
#pragma unroll
    for (int ks = 0; ks < 4; ++ks) {
        const int kb = ks * 32 + kh * 16;
        // A fragment: e=0..15 -> K=kb+e (contiguous, matches B)
        v8bf lo1 = *reinterpret_cast<const v8bf*>(a1row + kb);
        v8bf hi1 = *reinterpret_cast<const v8bf*>(a1row + kb + 8);
#pragma unroll
        for (int i = 0; i < 8; ++i) { aF1[ks][i] = lo1[i]; aF1[ks][i + 8] = hi1[i]; }
        // B fragment: K = 32*ks + 16*kh + e, contiguous in column-major W
        bF1[ks] = *reinterpret_cast<const v16bf*>(Wt1 + (size_t)col * HDIM + ks * 32 + kh * 16);
        if constexpr (HAS2) {
            v8bf lo2 = *reinterpret_cast<const v8bf*>(a2row + kb);
            v8bf hi2 = *reinterpret_cast<const v8bf*>(a2row + kb + 8);
#pragma unroll
            for (int i = 0; i < 8; ++i) { aF2[ks][i] = lo2[i]; aF2[ks][i + 8] = hi2[i]; }
            bF2[ks] = *reinterpret_cast<const v16bf*>(Wt2 + (size_t)col * HDIM + ks * 32 + kh * 16);
        }
    }

    // ---- WMMA chain (back-to-back accumulate) ------------------------------
    v8f acc = {};
#pragma unroll
    for (int ks = 0; ks < 4; ++ks) {
        acc = __builtin_amdgcn_wmma_f32_16x16x32_bf16(false, aF1[ks], false, bF1[ks],
                                                      (short)0, acc, false, false);
        if constexpr (HAS2) {
            acc = __builtin_amdgcn_wmma_f32_16x16x32_bf16(false, aF2[ks], false, bF2[ks],
                                                          (short)0, acc, false, false);
        }
    }

    const float bv = bias[col];
    const size_t obase = (size_t)(row0 + kh * 8) * HDIM + col;  // C VGPR r -> row r+8*kh
#pragma unroll
    for (int r = 0; r < 8; ++r) {
        float v = acc[r] + bv;
        if constexpr (RELU) v = fmaxf(v, 0.f);
        out[obase + (size_t)r * HDIM] = v;
    }
}

// ------------------------------------------------------------------ orchestration
extern "C" void kernel_launch(void* const* d_in, const int* in_sizes, int n_in,
                              void* d_out, int out_size, void* d_ws, size_t ws_size,
                              hipStream_t stream) {
    const int N = 100000;

    const float* x_A  = (const float*)d_in[0];
    const int* edge_r0 = (const int*)d_in[2];
    const int* edge_r1 = (const int*)d_in[3];
    const float* wl0 = (const float*)d_in[4];
    const float* bl0 = (const float*)d_in[5];
    const float* w00 = (const float*)d_in[6];
    const float* b00 = (const float*)d_in[7];
    const float* w10 = (const float*)d_in[8];
    const float* b10 = (const float*)d_in[9];
    const float* wl1 = (const float*)d_in[10];
    const float* bl1 = (const float*)d_in[11];
    const float* w01 = (const float*)d_in[12];
    const float* b01 = (const float*)d_in[13];
    const float* w11 = (const float*)d_in[14];
    const float* b11 = (const float*)d_in[15];
    const float* w_out = (const float*)d_in[16];
    const float* b_out = (const float*)d_in[17];

    const int E0 = in_sizes[2] / 2;
    const int E1 = in_sizes[3] / 2;

    // workspace: agg (51.2MB) | 5 bf16 weights | 2 bias vectors
    char* ws = (char*)d_ws;
    float* agg = (float*)ws;
    char* wbase = ws + (size_t)N * HDIM * sizeof(float);
    unsigned short* wlT1  = (unsigned short*)(wbase + 0 * 16384 * 2);
    unsigned short* wsT1  = (unsigned short*)(wbase + 1 * 16384 * 2);
    unsigned short* wlT0  = (unsigned short*)(wbase + 2 * 16384 * 2);
    unsigned short* wsT0  = (unsigned short*)(wbase + 3 * 16384 * 2);
    unsigned short* woutT = (unsigned short*)(wbase + 4 * 16384 * 2);
    float* bias1 = (float*)(wbase + 5 * 16384 * 2);
    float* bias2 = bias1 + HDIM;

    // h ping buffer lives in d_out (in-place GEMM is safe: each block reads its
    // 16 rows into LDS, barriers, then writes the same rows)
    float* h = (float*)d_out;

    prep_weights<<<64, 256, 0, stream>>>(wl1, w01, w11, bl1, b01, b11,
                                         wl0, w00, w10, bl0, b00, b10, w_out,
                                         wlT1, wsT1, wlT0, wsT0, woutT, bias1, bias2);

    const long long n4 = (long long)N * HDIM / 4;
    const unsigned zgrid = (unsigned)((n4 + 255) / 256);
    const unsigned ggrid = (unsigned)(N / 16);   // 6250, exact

    // ---- layer 1: (B, r1, A) ----
    zero_f32x4<<<zgrid, 256, 0, stream>>>(agg, n4);
    {
        long long t = (long long)E1 * 32;
        scatter_add<<<(unsigned)((t + 255) / 256), 256, 0, stream>>>(edge_r1, E1, x_A, agg);
    }
    gemm_tile<true, true><<<ggrid, 256, 0, stream>>>(agg, wlT1, x_A, wsT1, bias1, h);

    // ---- layer 2: (A, r0, A) ----
    zero_f32x4<<<zgrid, 256, 0, stream>>>(agg, n4);
    {
        long long t = (long long)E0 * 32;
        scatter_add<<<(unsigned)((t + 255) / 256), 256, 0, stream>>>(edge_r0, E0, h, agg);
    }
    gemm_tile<true, true><<<ggrid, 256, 0, stream>>>(agg, wlT0, h, wsT0, bias2, h);

    // ---- output projection ----
    gemm_tile<false, false><<<ggrid, 256, 0, stream>>>(h, woutT, nullptr, nullptr,
                                                       b_out, (float*)d_out);
}
